// WindowAttention_85701777424468
// MI455X (gfx1250) — matrix operand order repaired and checked by reading, not compiled, not run on hardware
//
#include <hip/hip_runtime.h>
#include <hip/hip_bf16.h>
#include <cstdint>

typedef __attribute__((ext_vector_type(16))) _Float16 v16h;
typedef __attribute__((ext_vector_type(8)))  _Float16 v8h;
typedef __attribute__((ext_vector_type(8)))  float    v8f;
typedef unsigned int v4u  __attribute__((ext_vector_type(4)));
typedef int          v8i_ __attribute__((ext_vector_type(8)));
typedef int          v4i_ __attribute__((ext_vector_type(4)));

#define WS 8
#define NH 8
#define HD 32
#define NG 4
#define GC 64
#define CDIM 256
#define NTOK 64
#define SCALE_ATTN 0.1767766952966369f

#if __has_builtin(__builtin_amdgcn_tensor_load_to_lds) && \
    __has_builtin(__builtin_amdgcn_s_wait_tensorcnt)
#define USE_TDM 1
#else
#define USE_TDM 0
#endif

__device__ __forceinline__ v16h mk16(v8h lo, v8h hi) {
    v16h r;
#pragma unroll
    for (int i = 0; i < 8; ++i) { r[i] = lo[i]; r[i + 8] = hi[i]; }
    return r;
}

// ---------------------------------------------------------------------------
// Fused per-(batch,head) attention: one wave computes S = QK^T (16 WMMA),
// row softmax (+ optional rel-pos bias), then P*V (16 WMMA).
// Token-major I/O: X[(b*64+n)*256 + h*32 + d]. Output f16.
// ---------------------------------------------------------------------------
template <typename TIN>
__global__ __launch_bounds__(32) void k_attn(const TIN* __restrict__ Q,
                                             const TIN* __restrict__ K,
                                             const TIN* __restrict__ V,
                                             const float* __restrict__ rpb,
                                             _Float16* __restrict__ O,
                                             int use_bias) {
    __shared__ __align__(16) _Float16 Qs[NTOK * HD];
    __shared__ __align__(16) _Float16 Ks[NTOK * HD];
    __shared__ __align__(16) _Float16 Vt[HD * NTOK];   // transposed: Vt[d][n]
    __shared__ __align__(16) float    S[NTOK * NTOK];
    __shared__ __align__(16) _Float16 P[NTOK * NTOK];

    const int bh = blockIdx.x;
    const int b = bh >> 3, h = bh & 7;
    const int lane = threadIdx.x;

    for (int i = lane; i < NTOK * HD; i += 32) {
        const int n = i >> 5, d = i & 31;
        const size_t gi = ((size_t)(b * NTOK + n)) * CDIM + h * HD + d;
        Qs[i] = (_Float16)((float)Q[gi] * SCALE_ATTN);
        Ks[i] = (_Float16)(float)K[gi];
        Vt[d * NTOK + n] = (_Float16)(float)V[gi];
    }
    __syncthreads();

    const int lm = lane & 15, hi = lane >> 4;

    // ---- S = Q * K^T : 4x4 tiles, K-dim = 32 = one WMMA each ----
    for (int ti = 0; ti < 4; ++ti) {
        for (int tj = 0; tj < 4; ++tj) {
            const _Float16* ap = &Qs[(ti * 16 + lm) * HD];
            v16h a = mk16(*(const v8h*)(ap + 8 * hi), *(const v8h*)(ap + 16 + 8 * hi));
            const _Float16* bp = &Ks[(tj * 16 + lm) * HD + 8 * hi];
            v16h bb = mk16(*(const v8h*)bp, *(const v8h*)(bp + 16));
            v8f c = {};
            c = __builtin_amdgcn_wmma_f32_16x16x32_f16(false, a, false, bb,
                                                       (short)0, c, false, false);
            const int col = tj * 16 + lm;
#pragma unroll
            for (int r = 0; r < 8; ++r)
                S[(ti * 16 + r + 8 * hi) * NTOK + col] = c[r];
        }
    }
    __syncthreads();

    // ---- row softmax (+ rel-pos bias) -> P (f16) ----
    for (int rr = 0; rr < 2; ++rr) {
        const int n = lane * 2 + rr;
        const int ny = n >> 3, nx = n & 7;
        float mx = -3.0e38f;
        for (int m = 0; m < NTOK; ++m) {
            float bv = 0.0f;
            if (use_bias) {
                const int dy = ny - (m >> 3) + (WS - 1);
                const int dx = nx - (m & 7) + (WS - 1);
                bv = rpb[(dy * (2 * WS - 1) + dx) * NH + h];
            }
            const float s = S[n * NTOK + m] + bv;
            S[n * NTOK + m] = s;
            mx = s > mx ? s : mx;
        }
        float sum = 0.0f;
        for (int m = 0; m < NTOK; ++m) {
            const float e = __expf(S[n * NTOK + m] - mx);
            sum += e;
            S[n * NTOK + m] = e;
        }
        const float inv = 1.0f / sum;
        for (int m = 0; m < NTOK; ++m)
            P[n * NTOK + m] = (_Float16)(S[n * NTOK + m] * inv);
    }
    __syncthreads();

    // ---- O = P * V : 4x2 tiles, K-dim 64 = 2 WMMA each ----
    for (int ti = 0; ti < 4; ++ti) {
        for (int tj = 0; tj < 2; ++tj) {
            v8f c = {};
#pragma unroll
            for (int ks = 0; ks < 2; ++ks) {
                const _Float16* ap = &P[(ti * 16 + lm) * NTOK + ks * 32];
                v16h a = mk16(*(const v8h*)(ap + 8 * hi), *(const v8h*)(ap + 16 + 8 * hi));
                const _Float16* bp = &Vt[(tj * 16 + lm) * NTOK + ks * 32 + 8 * hi];
                v16h bb = mk16(*(const v8h*)bp, *(const v8h*)(bp + 16));
                c = __builtin_amdgcn_wmma_f32_16x16x32_f16(false, a, false, bb,
                                                           (short)0, c, false, false);
            }
            const int d = tj * 16 + lm;
#pragma unroll
            for (int r = 0; r < 8; ++r) {
                const int m = ti * 16 + r + 8 * hi;
                O[((size_t)(b * NTOK + m)) * CDIM + h * HD + d] = (_Float16)c[r];
            }
        }
    }
}

// ---------------------------------------------------------------------------
// Y[M x 256] = A[M x 256](f16) @ W^T (f16, oc x ic row-major) + bias(f32).
// Block: 256 threads (8 waves), 64 rows per block. A-tile is one contiguous
// 32KB global block -> staged into LDS via TDM tensor_load_to_lds (6-arg
// clang-23 form), else vectorized copy. 64 WMMA per wave.
// out_mode: 0 -> f16 token-major Yh; 1 -> f32 channel-major Yf (final output)
// ---------------------------------------------------------------------------
__global__ __launch_bounds__(256) void k_gemm(const _Float16* __restrict__ A,
                                              const _Float16* __restrict__ W,
                                              const float* __restrict__ bias,
                                              _Float16* __restrict__ Yh,
                                              float* __restrict__ Yf,
                                              int out_mode) {
    __shared__ __align__(16) _Float16 As[64 * CDIM];
    const int row0 = blockIdx.x * 64;
    const int tid = threadIdx.x;

#if USE_TDM
    if ((tid >> 5) == 0) {  // wave 0 issues one TDM descriptor for the 32KB tile
        const uint64_t gaddr = (uint64_t)(uintptr_t)(A + (size_t)row0 * CDIM);
        const unsigned ldsOff = (unsigned)(uintptr_t)&As[0];  // low 32 bits = LDS offset
        v4u g0;
        g0[0] = 1u;                                   // count=1, user descriptor
        g0[1] = ldsOff;                               // lds_addr
        g0[2] = (unsigned)gaddr;                      // global_addr[31:0]
        g0[3] = (unsigned)((gaddr >> 32) & 0x01FFFFFFu) | 0x80000000u; // addr[56:32], type=2
        v8i_ g1;
        g1[0] = 0x00010000;       // data_size=1 (2 bytes); no multicast/pad/iterate
        g1[1] = 0x40000000;       // tensor_dim0 = 16384 (low16 << 16)
        g1[2] = 0x00010000;       // tensor_dim1 = 1
        g1[3] = 0x40000000;       // tile_dim0 = 16384
        g1[4] = 1;                // tile_dim1 = 1
        g1[5] = 16384;            // tensor_dim0_stride
        g1[6] = 0;
        g1[7] = 0;
        v4i_ gz4 = {0, 0, 0, 0};
        v8i_ gz8 = {0, 0, 0, 0, 0, 0, 0, 0};
        __builtin_amdgcn_tensor_load_to_lds(g0, g1, gz4, gz4, gz8, 0);
        __builtin_amdgcn_s_wait_tensorcnt(0);
    }
#else
    {
        const uint4* src = (const uint4*)(A + (size_t)row0 * CDIM);
        uint4* dst = (uint4*)As;
        for (int i = tid; i < (64 * CDIM) / 8; i += 256) dst[i] = src[i];
    }
#endif
    __syncthreads();

    const int wave = tid >> 5, lane = tid & 31;
    const int lm = lane & 15, hi = lane >> 4;

    v8f acc[4][2];
#pragma unroll
    for (int ti = 0; ti < 4; ++ti)
#pragma unroll
        for (int tj = 0; tj < 2; ++tj) acc[ti][tj] = (v8f){};

    for (int ks = 0; ks < 8; ++ks) {
        const int k0 = ks * 32;
        v16h a[4];
#pragma unroll
        for (int ti = 0; ti < 4; ++ti) {
            const _Float16* ap = &As[(ti * 16 + lm) * CDIM + k0];
            a[ti] = mk16(*(const v8h*)(ap + 8 * hi), *(const v8h*)(ap + 16 + 8 * hi));
        }
#pragma unroll
        for (int tj = 0; tj < 2; ++tj) {
            const int n = (wave * 2 + tj) * 16 + lm;
            const _Float16* wp = &W[(size_t)n * CDIM + k0 + 8 * hi];
            v16h bb = mk16(*(const v8h*)wp, *(const v8h*)(wp + 16));  // two b128 loads
#pragma unroll
            for (int ti = 0; ti < 4; ++ti)
                acc[ti][tj] = __builtin_amdgcn_wmma_f32_16x16x32_f16(
                    false, a[ti], false, bb, (short)0, acc[ti][tj], false, false);
        }
    }

#pragma unroll
    for (int ti = 0; ti < 4; ++ti)
#pragma unroll
        for (int tj = 0; tj < 2; ++tj) {
            const int col = (wave * 2 + tj) * 16 + lm;
            const float bv = bias[col];
#pragma unroll
            for (int r = 0; r < 8; ++r) {
                const int m = row0 + ti * 16 + r + 8 * hi;
                const float val = acc[ti][tj][r] + bv;
                if (out_mode == 0) {
                    Yh[(size_t)m * CDIM + col] = (_Float16)val;
                } else {
                    const int bb_ = m >> 6, tok = m & 63;
                    Yf[((size_t)bb_ * CDIM + col) * NTOK + tok] = val;
                }
            }
        }
}

// ---------------------------------------------------------------------------
// f32 -> f16 weight conversion (once per launch; 256KB -> 128KB, L2 resident)
// ---------------------------------------------------------------------------
__global__ __launch_bounds__(256) void k_cvt(const float* __restrict__ src,
                                             _Float16* __restrict__ dst, int n) {
    const int i = blockIdx.x * 256 + threadIdx.x;
    if (i < n) dst[i] = (_Float16)src[i];
}

// ---------------------------------------------------------------------------
// Offset subnet + grid-sample, fused per (batch, group). One thread per
// spatial position: depthwise 5x5 conv -> LN -> GELU -> pointwise(2) ->
// tanh*range -> pos = ref + off -> bilinear sample of x (f16 in, f16 out).
// ---------------------------------------------------------------------------
__global__ __launch_bounds__(64) void k_off_sample(const _Float16* __restrict__ qc,
                                                   const _Float16* __restrict__ x,
                                                   const float* __restrict__ dww,
                                                   const float* __restrict__ dwb,
                                                   const float* __restrict__ lng,
                                                   const float* __restrict__ lnb,
                                                   const float* __restrict__ pww,
                                                   _Float16* __restrict__ xs,
                                                   float* __restrict__ pos_out,
                                                   float* __restrict__ ref_out) {
    __shared__ float Xl[64][GC];
    __shared__ float T[64][GC];
    const int bg = blockIdx.x;
    const int b = bg >> 2, g = bg & 3;
    const int n = threadIdx.x;
    const int py = n >> 3, px = n & 7;

    for (int c = 0; c < GC; ++c)
        Xl[n][c] = (float)qc[((size_t)(b * NTOK + n)) * CDIM + g * GC + c];
    __syncthreads();

    float mean = 0.0f, msq = 0.0f;
    for (int c = 0; c < GC; ++c) {
        float acc = dwb[c];
#pragma unroll
        for (int dy = 0; dy < 5; ++dy) {
            const int yy = py + dy - 2;
#pragma unroll
            for (int dx = 0; dx < 5; ++dx) {
                const int xx = px + dx - 2;
                if (yy >= 0 && yy < 8 && xx >= 0 && xx < 8)
                    acc += dww[c * 25 + dy * 5 + dx] * Xl[yy * 8 + xx][c];
            }
        }
        T[n][c] = acc;
        mean += acc;
        msq += acc * acc;
    }
    mean *= (1.0f / GC);
    const float var = msq * (1.0f / GC) - mean * mean;
    const float rstd = rsqrtf(var + 1e-5f);

    float oy = 0.0f, ox = 0.0f;
    for (int c = 0; c < GC; ++c) {
        const float u = (T[n][c] - mean) * rstd * lng[c] + lnb[c];
        const float ge = 0.5f * u * (1.0f + erff(u * 0.70710678118654752f));
        oy += pww[c] * ge;
        ox += pww[GC + c] * ge;
    }
    const float offy = tanhf(oy) * 0.25f;   // (1/8)*ORF(2.0)
    const float offx = tanhf(ox) * 0.25f;
    const float refy = ((py + 0.5f) * 0.125f) * 2.0f - 1.0f;
    const float refx = ((px + 0.5f) * 0.125f) * 2.0f - 1.0f;
    const float posy = refy + offy, posx = refx + offx;

    const size_t o = ((size_t)bg * NTOK + n) * 2;
    pos_out[o] = posy; pos_out[o + 1] = posx;
    ref_out[o] = refy; ref_out[o + 1] = refx;

    const float gx = (posx + 1.0f) * 0.5f * 7.0f;
    const float gy = (posy + 1.0f) * 0.5f * 7.0f;
    const float x0f = floorf(gx), y0f = floorf(gy);
    const int x0 = (int)x0f, y0 = (int)y0f;
    const float wx1 = gx - x0f, wy1 = gy - y0f;

    int idx4[4];
    float wv[4];
    const int xi[4] = {x0, x0 + 1, x0, x0 + 1};
    const int yi[4] = {y0, y0, y0 + 1, y0 + 1};
    const float w4[4] = {(1.0f - wx1) * (1.0f - wy1), wx1 * (1.0f - wy1),
                         (1.0f - wx1) * wy1, wx1 * wy1};
#pragma unroll
    for (int kk = 0; kk < 4; ++kk) {
        const bool valid = (xi[kk] >= 0) && (xi[kk] <= 7) && (yi[kk] >= 0) && (yi[kk] <= 7);
        const int xc = xi[kk] < 0 ? 0 : (xi[kk] > 7 ? 7 : xi[kk]);
        const int yc = yi[kk] < 0 ? 0 : (yi[kk] > 7 ? 7 : yi[kk]);
        idx4[kk] = yc * 8 + xc;
        wv[kk] = valid ? w4[kk] : 0.0f;
    }
    for (int c = 0; c < GC; ++c) {
        float v = 0.0f;
#pragma unroll
        for (int kk = 0; kk < 4; ++kk)
            v += wv[kk] * (float)x[((size_t)(b * NTOK + idx4[kk])) * CDIM + g * GC + c];
        xs[((size_t)(b * NTOK + n)) * CDIM + g * GC + c] = (_Float16)v;
    }
}

extern "C" void kernel_launch(void* const* d_in, const int* in_sizes, int n_in,
                              void* d_out, int out_size, void* d_ws, size_t ws_size,
                              hipStream_t stream) {
    const float* q      = (const float*)d_in[0];
    const float* k      = (const float*)d_in[1];
    const float* v      = (const float*)d_in[2];
    const float* rpb    = (const float*)d_in[3];
    const float* proj_w = (const float*)d_in[4];
    const float* proj_b = (const float*)d_in[5];
    const float* dww    = (const float*)d_in[6];
    const float* dwb    = (const float*)d_in[7];
    const float* lng    = (const float*)d_in[8];
    const float* lnb    = (const float*)d_in[9];
    const float* pww    = (const float*)d_in[10];
    const float* pq_w   = (const float*)d_in[11];
    const float* pq_b   = (const float*)d_in[12];
    const float* pk_w   = (const float*)d_in[13];
    const float* pk_b   = (const float*)d_in[14];
    const float* pv_w   = (const float*)d_in[15];
    const float* pv_b   = (const float*)d_in[16];
    const float* po_w   = (const float*)d_in[17];
    const float* po_b   = (const float*)d_in[18];

    const int B = 2048;
    const size_t NELEM = (size_t)B * NTOK * CDIM;   // 33,554,432 elements
    const int WN = CDIM * CDIM;                      // 65,536 per weight

    // f16 scratch: 4 activation buffers + 5 converted weight matrices
    _Float16* h0 = (_Float16*)d_ws;      // x1 -> x_sampled -> out2
    _Float16* h1 = h0 + NELEM;           // x  -> k2
    _Float16* h2 = h1 + NELEM;           // qc
    _Float16* h3 = h2 + NELEM;           // v2
    _Float16* wh = h3 + NELEM;           // 5 x 65536 halves
    _Float16* w_proj = wh;
    _Float16* w_pq   = wh + WN;
    _Float16* w_pk   = wh + 2 * WN;
    _Float16* w_pv   = wh + 3 * WN;
    _Float16* w_po   = wh + 4 * WN;

    float* y       = (float*)d_out;                       // (B,C,HW) flat
    float* pos_out = y + NELEM;                           // (B,4,8,8,2)
    float* ref_out = pos_out + (size_t)B * NG * NTOK * 2; // (B,4,8,8,2)

    const int nBH = B * NH;               // 16384 wave-blocks
    const int nGemmBlk = (B * NTOK) / 64; // 2048
    const int nCvtBlk = WN / 256;

    // 0) convert weights to f16 (L2-resident afterwards)
    k_cvt<<<nCvtBlk, 256, 0, stream>>>(proj_w, w_proj, WN);
    k_cvt<<<nCvtBlk, 256, 0, stream>>>(pq_w,   w_pq,   WN);
    k_cvt<<<nCvtBlk, 256, 0, stream>>>(pk_w,   w_pk,   WN);
    k_cvt<<<nCvtBlk, 256, 0, stream>>>(pv_w,   w_pv,   WN);
    k_cvt<<<nCvtBlk, 256, 0, stream>>>(po_w,   w_po,   WN);

    // 1) windowed MHSA with rel-pos bias -> h0 (x1, f16 token-major)
    k_attn<float><<<nBH, 32, 0, stream>>>(q, k, v, rpb, h0, 1);
    // 2) proj -> h1 (x)
    k_gemm<<<nGemmBlk, 256, 0, stream>>>(h0, w_proj, proj_b, h1, nullptr, 0);
    // 3) pq -> h2 (qc)
    k_gemm<<<nGemmBlk, 256, 0, stream>>>(h1, w_pq, pq_b, h2, nullptr, 0);
    // 4) offset subnet + grid sample: qc, x -> xs (h0), pos/ref -> d_out tail
    k_off_sample<<<B * NG, 64, 0, stream>>>(h2, h1, dww, dwb, lng, lnb, pww,
                                            h0, pos_out, ref_out);
    // 5) pk -> h1 (k2), 6) pv -> h3 (v2)
    k_gemm<<<nGemmBlk, 256, 0, stream>>>(h0, w_pk, pk_b, h1, nullptr, 0);
    k_gemm<<<nGemmBlk, 256, 0, stream>>>(h0, w_pv, pv_b, h3, nullptr, 0);
    // 7) deformable attention (no bias): qc x k2 -> softmax -> *v2 -> h0
    k_attn<_Float16><<<nBH, 32, 0, stream>>>(h2, h1, h3, nullptr, h0, 0);
    // 8) po -> y, f32 channel-major (B,C,HW) to match the raw reshape
    k_gemm<<<nGemmBlk, 256, 0, stream>>>(h0, w_po, po_b, nullptr, y, 1);
}
